// LTC1_89773406421389
// MI455X (gfx1250) — matrix operand order repaired and checked by reading, not compiled, not run on hardware
//
#include <hip/hip_runtime.h>
#include <hip/hip_bf16.h>
#include <stdint.h>

// ---------------------------------------------------------------------------
// MI455X (gfx1250) implementation of CNN + LTC forward.
// Convs:  implicit-GEMM via v_wmma_f32_16x16x32_f16 (f16 in, f32 accum),
//         A-tile (weights) double-buffered in LDS via async global->LDS copies
//         (ASYNCcnt) with a synchronous ds_store fallback.
// LTC:    one wave32 per sample, lane = neuron, __shfl for recurrent state.
// ---------------------------------------------------------------------------

typedef __attribute__((ext_vector_type(16))) _Float16 v16h;
typedef __attribute__((ext_vector_type(8)))  float    v8f;
typedef __attribute__((ext_vector_type(2)))  int      v2i;

#define DEV __device__ __forceinline__

DEV float sigmoidf_(float x) { return 1.0f / (1.0f + __expf(-x)); }
DEV float softplusf_(float x) { return logf(1.0f + __expf(x)); }

template <typename T> DEV _Float16 to_h(T x) { return (_Float16)(float)x; }

// ---- gfx1250 async global->LDS staging (guarded; fallback = sync + ds_store)
#if defined(__has_builtin)
#if __has_builtin(__builtin_amdgcn_global_load_async_to_lds_b64)
#define HAVE_ASYNC_LDS 1
#endif
#endif

DEV void stage_b64(const _Float16* __restrict__ g, _Float16* l) {
#if defined(HAVE_ASYNC_LDS)
  // signature (from compiler diagnostic): (v2i as1* src, v2i as3* dst, imm off, imm cpol)
  __builtin_amdgcn_global_load_async_to_lds_b64(
      (__attribute__((address_space(1))) v2i*)g,
      (__attribute__((address_space(3))) v2i*)l, 0, 0);
#else
  *(uint64_t*)l = *(const uint64_t*)g;   // global_load_b64 + ds_store_b64
#endif
}

DEV void wait_stage() {
#if defined(HAVE_ASYNC_LDS)
#if __has_builtin(__builtin_amdgcn_s_wait_asynccnt)
  __builtin_amdgcn_s_wait_asynccnt(0);
#else
  asm volatile("s_wait_asynccnt 0x0" ::: "memory");
#endif
#endif
}

// ---------------------------------------------------------------------------
// Prep: fold bias/BN into scale+shift, f16 weights (K zero-padded to mult. of
// 32 so A-staging is unconditional aligned b64), LTC parameter tables.
// ---------------------------------------------------------------------------
struct LtcParams {
  const float *conv1_w, *conv1_b, *conv2_w, *conv2_b, *bn2_g, *bn2_b, *bn2_m, *bn2_v;
  const float *conv3_w, *conv3_b, *conv4_w, *conv4_b, *bn4_g, *bn4_b, *bn4_m, *bn4_v;
  const float *sensory_w, *sensory_erev, *sensory_mask;
  const float *w, *erev, *mask, *gleak, *cm;
  _Float16 *w1h, *w2h, *w3h, *w4h;
  float *scale1, *shift1, *scale2, *shift2, *scale3, *shift3, *scale4, *shift4;
  float *sens_weff, *sens_we, *w_eff, *w_we, *gl, *cmt;
};

__global__ void prep_kernel(LtcParams P) {
  const int id = blockIdx.x * blockDim.x + threadIdx.x;
  if (id < 512) {                         // conv1: 16 x (9 -> pad 32)
    const int m = id / 32, k = id % 32;
    P.w1h[id] = (k < 9) ? (_Float16)P.conv1_w[m * 9 + k] : (_Float16)0.0f;
  }
  if (id < 5120) {                        // conv2: 32 x (144 -> pad 160)
    const int m = id / 160, k = id % 160;
    P.w2h[id] = (k < 144) ? (_Float16)P.conv2_w[m * 144 + k] : (_Float16)0.0f;
  }
  if (id < 51200)  P.w3h[id] = (_Float16)P.conv3_w[id];   // K=800  (exact)
  if (id < 204800) P.w4h[id] = (_Float16)P.conv4_w[id];   // K=1600 (exact)
  if (id < 16) { P.scale1[id] = 1.0f; P.shift1[id] = P.conv1_b[id]; }
  if (id < 32) {
    const float inv = P.bn2_g[id] * rsqrtf(P.bn2_v[id] + 1e-5f);
    P.scale2[id] = inv;
    P.shift2[id] = P.conv2_b[id] * inv + P.bn2_b[id] - P.bn2_m[id] * inv;
  }
  if (id < 64) { P.scale3[id] = 1.0f; P.shift3[id] = P.conv3_b[id]; }
  if (id < 128) {
    const float inv = P.bn4_g[id] * rsqrtf(P.bn4_v[id] + 1e-5f);
    P.scale4[id] = inv;
    P.shift4[id] = P.conv4_b[id] * inv + P.bn4_b[id] - P.bn4_m[id] * inv;
  }
  if (id < 304) {                         // sensory synapses: 16 x 19
    const float sw = softplusf_(P.sensory_w[id]) * P.sensory_mask[id];
    P.sens_weff[id] = sw;
    P.sens_we[id]   = sw * P.sensory_erev[id];
  }
  if (id < 361) {                         // recurrent synapses: 19 x 19
    const float sw = softplusf_(P.w[id]) * P.mask[id];
    P.w_eff[id] = sw;
    P.w_we[id]  = sw * P.erev[id];
  }
  if (id < 19) {
    P.gl[id]  = softplusf_(P.gleak[id]);
    P.cmt[id] = softplusf_(P.cm[id]) * 6.0f;   // cm / (1.0 / ODE_UNFOLDS)
  }
}

// ---------------------------------------------------------------------------
// Implicit-GEMM conv via WMMA.
//   M = OUT_C, N = batch*OUT_H*OUT_W, K = IN_C*KS*KS (weights padded to KP).
//   Per block: M_WAVES*16 oc rows staged in LDS per 32-wide K-step, double
//   buffered (async copy of step ks+1 overlaps B-gather + WMMA of step ks).
//   Fragment layouts per ISA 7.12.2 (16-bit A 16x32, B 32x16, f32 C/D 16x16):
//     A: lane L (m=L&15, hi=L>>4), vgpr r holds K = base,base+1,
//        base = (r<4 ? 2r : 16+2(r-4)) + 8*hi.
//     B: same K order as A (vgpr r holds K = base,base+1, same base);
//        column n = L&15.
//     D: vgpr r -> row (r + 8*hi), col (L&15).
// ---------------------------------------------------------------------------
template <int IN_C, int IN_H, int IN_W, int OUT_C, int OUT_H, int OUT_W,
          int KS, int STR, int PAD, int M_WAVES, int N_WAVES, typename Tin>
__global__ __launch_bounds__(32 * M_WAVES * N_WAVES)
void conv_wmma_kernel(const Tin* __restrict__ in, const _Float16* __restrict__ wgt,
                      const float* __restrict__ scale, const float* __restrict__ shift,
                      _Float16* __restrict__ out, int batch)
{
  constexpr int K = IN_C * KS * KS;
  constexpr int KSTEPS = (K + 31) / 32;
  constexpr int KP = KSTEPS * 32;                 // padded K (weight row stride)
  constexpr bool KCHK = (K % 32) != 0;            // B needs k<K predicate?
  constexpr int ROWS = M_WAVES * 16;              // oc rows per block
  constexpr int ELEMS = ROWS * 32;                // halves per K-step stage

  __shared__ _Float16 smA[2][ELEMS];              // double-buffered A tile

  const int tid  = threadIdx.x;
  const int lane = tid & 31;
  const int wave = tid >> 5;
  const int wm = wave / N_WAVES, wn = wave % N_WAVES;
  const int m0blk = blockIdx.y * ROWS;
  const int m0 = m0blk + wm * 16;
  const int n0 = (blockIdx.x * N_WAVES + wn) * 16;
  const int hi = lane >> 4;
  const int n  = n0 + (lane & 15);                // B/D column owned by this lane

  const int hw = OUT_H * OUT_W;
  int b = n / hw;
  const int p  = n % hw;
  const int oh = p / OUT_W, ow = p % OUT_W;
  const bool nvalid = (b < batch);
  if (!nvalid) b = 0;
  const int ih0 = oh * STR - PAD, iw0 = ow * STR - PAD;
  const Tin* __restrict__ inb = in + (long)b * IN_C * IN_H * IN_W;
  __builtin_prefetch(inb, 0, 0);                  // global_prefetch_b8

  // staging assignment: thread copies 4 halves (aligned b64) of the A tile
  const int se   = tid * 4;
  const int srow = se >> 5;                       // tile-local oc row
  const int skk  = se & 31;
  const bool scopy = (tid < ELEMS / 4);
  const _Float16* __restrict__ gsrc =
      wgt + (size_t)(m0blk + srow) * KP + skk;    // + ks*32 per step
  if (scopy) stage_b64(gsrc, &smA[0][se]);        // prologue: stage step 0

  v8f acc = {0.f, 0.f, 0.f, 0.f, 0.f, 0.f, 0.f, 0.f};
  for (int ks = 0; ks < KSTEPS; ++ks) {
    wait_stage();          // wave's outstanding async stages complete
    __syncthreads();       // all waves' stage of buf[ks&1] visible; prior reads done
    if (ks + 1 < KSTEPS && scopy)
      stage_b64(gsrc + (ks + 1) * 32, &smA[(ks + 1) & 1][se]);

    const int k0 = ks * 32;
    const _Float16* smrow = &smA[ks & 1][(wm * 16 + (lane & 15)) * 32];
    v16h Af, Bf;
#pragma unroll
    for (int r = 0; r < 8; ++r) {
      // A fragment from LDS (ds_load), layout base = (r<4?2r:16+2(r-4)) + 8*hi
      const int ka = ((r < 4) ? 2 * r : 16 + 2 * (r - 4)) + 8 * hi;
      Af[2 * r]     = smrow[ka];
      Af[2 * r + 1] = smrow[ka + 1];
      // B fragment: im2col gather from global (L2-resident activations)
      const int kb = k0 + ka;
#pragma unroll
      for (int e = 0; e < 2; ++e) {
        const int k = kb + e;
        _Float16 val = (_Float16)0.0f;
        if (nvalid && (!KCHK || k < K)) {
          const int c  = k / (KS * KS);
          const int rr = k % (KS * KS);
          const int ih = ih0 + rr / KS;
          const int iw = iw0 + rr % KS;
          if (ih >= 0 && ih < IN_H && iw >= 0 && iw < IN_W)
            val = to_h(inb[(c * IN_H + ih) * IN_W + iw]);
        }
        Bf[2 * r + e] = val;
      }
    }
    acc = __builtin_amdgcn_wmma_f32_16x16x32_f16(
        /*neg_a=*/false, Af, /*neg_b=*/false, Bf,
        /*c_mod=*/(short)0, acc, /*reuse_a=*/false, /*reuse_b=*/false);
  }

  if (nvalid) {
#pragma unroll
    for (int r = 0; r < 8; ++r) {
      const int oc = m0 + r + 8 * hi;
      float y = acc[r] * scale[oc] + shift[oc];
      y = fmaxf(y, 0.0f);                // all four convs are ReLU'd
      out[((long)(b * OUT_C + oc) * OUT_H + oh) * OUT_W + ow] = (_Float16)y;
    }
  }
}

// ---------------------------------------------------------------------------
// 2x2 maxpool [B,128,4,4] -> f32 seq [B,512]  (flat view == reshape(B,32,16))
// ---------------------------------------------------------------------------
__global__ void pool_kernel(const _Float16* __restrict__ in, float* __restrict__ seq, int batch) {
  const int idx = blockIdx.x * blockDim.x + threadIdx.x;
  if (idx >= batch * 512) return;
  const int b = idx / 512;
  const int r = idx % 512;
  const int c = r / 4, q = r % 4;
  const int ph = q / 2, pw = q % 2;
  const _Float16* base = in + (((long)(b * 128 + c) * 4 + ph * 2) * 4 + pw * 2);
  const float m = fmaxf(fmaxf((float)base[0], (float)base[1]),
                        fmaxf((float)base[4], (float)base[5]));
  seq[idx] = m;
}

// ---------------------------------------------------------------------------
// LTC cell: one wave32 per sample; lane j = neuron j (19 active).
// Recurrent state v broadcast across lanes with __shfl (lane-permute HW).
// ---------------------------------------------------------------------------
__global__ __launch_bounds__(256)
void ltc_kernel(const float* __restrict__ seq,
                const float* __restrict__ smu, const float* __restrict__ ssig,
                const float* __restrict__ sens_we, const float* __restrict__ sens_weff,
                const float* __restrict__ mu, const float* __restrict__ sigma,
                const float* __restrict__ w_we, const float* __restrict__ w_eff,
                const float* __restrict__ gl, const float* __restrict__ vleak,
                const float* __restrict__ cmt,
                const float* __restrict__ inw, const float* __restrict__ inb,
                const float* __restrict__ outw, const float* __restrict__ outb,
                float* __restrict__ out, int batch)
{
  const int lane = threadIdx.x & 31;
  const int samp = blockIdx.x * (blockDim.x >> 5) + (threadIdx.x >> 5);
  if (samp >= batch) return;
  const int j = (lane < 19) ? lane : 18;    // clamp idle lanes to a valid column
  const float glj = gl[j], cmj = cmt[j];
  const float glvl = glj * vleak[j];
  const float denc = cmj + glj;
  const float* s = seq + (long)samp * 512;
  float v = 0.0f;
  for (int t = 0; t < 32; ++t) {
    // sensory synapses for this timestep (depend only on the input)
    float wns = 0.0f, wds = 0.0f;
#pragma unroll
    for (int ss = 0; ss < 16; ++ss) {
      const float xv = s[t * 16 + ss] * inw[ss] + inb[ss];
      const float g = sigmoidf_((xv - smu[ss * 19 + j]) * ssig[ss * 19 + j]);
      wns += sens_we[ss * 19 + j] * g;
      wds += sens_weff[ss * 19 + j] * g;
    }
    // fused ODE solver, 6 unfolds
    for (int u = 0; u < 6; ++u) {
      float na = 0.0f, da = 0.0f;
#pragma unroll
      for (int i = 0; i < 19; ++i) {
        const float vi = __shfl(v, i, 32);
        const float g = sigmoidf_((vi - mu[i * 19 + j]) * sigma[i * 19 + j]);
        na += w_we[i * 19 + j] * g;
        da += w_eff[i * 19 + j] * g;
      }
      v = (cmj * v + glvl + na + wns) / (denc + da + wds + 1e-8f);
    }
  }
  if (lane < 2) out[samp * 2 + lane] = v * outw[lane] + outb[lane];
}

// ---------------------------------------------------------------------------
extern "C" void kernel_launch(void* const* d_in, const int* in_sizes, int n_in,
                              void* d_out, int out_size, void* d_ws, size_t ws_size,
                              hipStream_t stream) {
  const float* x            = (const float*)d_in[0];
  const float* conv1_w      = (const float*)d_in[1];
  const float* conv1_b      = (const float*)d_in[2];
  const float* conv2_w      = (const float*)d_in[3];
  const float* conv2_b      = (const float*)d_in[4];
  const float* bn2_g        = (const float*)d_in[5];
  const float* bn2_b        = (const float*)d_in[6];
  const float* bn2_m        = (const float*)d_in[7];
  const float* bn2_v        = (const float*)d_in[8];
  const float* conv3_w      = (const float*)d_in[9];
  const float* conv3_b      = (const float*)d_in[10];
  const float* conv4_w      = (const float*)d_in[11];
  const float* conv4_b      = (const float*)d_in[12];
  const float* bn4_g        = (const float*)d_in[13];
  const float* bn4_b        = (const float*)d_in[14];
  const float* bn4_m        = (const float*)d_in[15];
  const float* bn4_v        = (const float*)d_in[16];
  const float* input_w      = (const float*)d_in[17];
  const float* input_b      = (const float*)d_in[18];
  const float* sensory_mu   = (const float*)d_in[19];
  const float* sensory_sig  = (const float*)d_in[20];
  const float* sensory_w    = (const float*)d_in[21];
  const float* sensory_erev = (const float*)d_in[22];
  const float* sensory_mask = (const float*)d_in[23];
  const float* mu           = (const float*)d_in[24];
  const float* sigma        = (const float*)d_in[25];
  const float* w            = (const float*)d_in[26];
  const float* erev         = (const float*)d_in[27];
  const float* mask         = (const float*)d_in[28];
  const float* gleak        = (const float*)d_in[29];
  const float* vleak        = (const float*)d_in[30];
  const float* cm           = (const float*)d_in[31];
  const float* output_w     = (const float*)d_in[32];
  const float* output_b     = (const float*)d_in[33];
  float* out = (float*)d_out;

  const int batch = in_sizes[0] / (28 * 28);

  // ---- workspace layout (256B aligned) ----
  char* ws = (char*)d_ws;
  size_t off = 0;
  auto take = [&](size_t bytes) -> void* {
    off = (off + 255) & ~(size_t)255;
    void* p = ws + off;
    off += bytes;
    return p;
  };
  _Float16* w1h = (_Float16*)take(512 * 2);      // 16 x 32  (K padded)
  _Float16* w2h = (_Float16*)take(5120 * 2);     // 32 x 160 (K padded)
  _Float16* w3h = (_Float16*)take(51200 * 2);    // 64 x 800
  _Float16* w4h = (_Float16*)take(204800 * 2);   // 128 x 1600
  float* scale1 = (float*)take(16 * 4);   float* shift1 = (float*)take(16 * 4);
  float* scale2 = (float*)take(32 * 4);   float* shift2 = (float*)take(32 * 4);
  float* scale3 = (float*)take(64 * 4);   float* shift3 = (float*)take(64 * 4);
  float* scale4 = (float*)take(128 * 4);  float* shift4 = (float*)take(128 * 4);
  float* sens_weff = (float*)take(304 * 4);
  float* sens_we   = (float*)take(304 * 4);
  float* w_eff     = (float*)take(361 * 4);
  float* w_we      = (float*)take(361 * 4);
  float* glp       = (float*)take(19 * 4);
  float* cmtp      = (float*)take(19 * 4);
  off = (off + 255) & ~(size_t)255;
  const size_t actBase = off;
  const size_t A1 = (size_t)batch * 16 * 26 * 26 * 2;   // 88.6 MB @ B=4096
  const size_t A3 = (size_t)batch * 64 * 7 * 7 * 2;     // 25.7 MB
  const size_t A4 = (size_t)batch * 128 * 4 * 4 * 2;    // 16.8 MB
  _Float16* act1 = (_Float16*)(ws + actBase);           // conv1 out
  _Float16* act2 = (_Float16*)(ws + actBase + A1);      // conv2 out
  _Float16* act3 = (_Float16*)(ws + actBase);           // conv3 out (reuses act1)
  _Float16* act4 = (_Float16*)(ws + actBase + A3);      // conv4 out
  float*    seqp = (float*)(ws + actBase + A3 + A4);    // pooled seq, f32
  (void)ws_size;  // peak usage ~141 MB

  // ---- 0) parameter prep ----
  LtcParams P;
  P.conv1_w = conv1_w; P.conv1_b = conv1_b; P.conv2_w = conv2_w; P.conv2_b = conv2_b;
  P.bn2_g = bn2_g; P.bn2_b = bn2_b; P.bn2_m = bn2_m; P.bn2_v = bn2_v;
  P.conv3_w = conv3_w; P.conv3_b = conv3_b; P.conv4_w = conv4_w; P.conv4_b = conv4_b;
  P.bn4_g = bn4_g; P.bn4_b = bn4_b; P.bn4_m = bn4_m; P.bn4_v = bn4_v;
  P.sensory_w = sensory_w; P.sensory_erev = sensory_erev; P.sensory_mask = sensory_mask;
  P.w = w; P.erev = erev; P.mask = mask; P.gleak = gleak; P.cm = cm;
  P.w1h = w1h; P.w2h = w2h; P.w3h = w3h; P.w4h = w4h;
  P.scale1 = scale1; P.shift1 = shift1; P.scale2 = scale2; P.shift2 = shift2;
  P.scale3 = scale3; P.shift3 = shift3; P.scale4 = scale4; P.shift4 = shift4;
  P.sens_weff = sens_weff; P.sens_we = sens_we; P.w_eff = w_eff; P.w_we = w_we;
  P.gl = glp; P.cmt = cmtp;
  prep_kernel<<<(204800 + 255) / 256, 256, 0, stream>>>(P);

  // ---- 1) conv1 3x3 s1 p0: [B,1,28,28] -> [B,16,26,26], ReLU ----
  {
    const int N = batch * 26 * 26;
    dim3 grid((N + 127) / 128, 1);   // 1 M-wave x 8 N-waves -> 16x128 tile
    conv_wmma_kernel<1, 28, 28, 16, 26, 26, 3, 1, 0, 1, 8, float>
        <<<grid, 256, 0, stream>>>(x, w1h, scale1, shift1, act1, batch);
  }
  // ---- 2) conv2 3x3 s2 p2 + BN + ReLU: -> [B,32,14,14] ----
  {
    const int N = batch * 14 * 14;
    dim3 grid((N + 63) / 64, 1);     // 2 M-waves x 4 N-waves -> 32x64 tile
    conv_wmma_kernel<16, 26, 26, 32, 14, 14, 3, 2, 2, 2, 4, _Float16>
        <<<grid, 256, 0, stream>>>(act1, w2h, scale2, shift2, act2, batch);
  }
  // ---- 3) conv3 5x5 s2 p2 + ReLU: -> [B,64,7,7] ----
  {
    const int N = batch * 7 * 7;
    dim3 grid((N + 63) / 64, 2);
    conv_wmma_kernel<32, 14, 14, 64, 7, 7, 5, 2, 2, 2, 4, _Float16>
        <<<grid, 256, 0, stream>>>(act2, w3h, scale3, shift3, act3, batch);
  }
  // ---- 4) conv4 5x5 s2 p2 + BN + ReLU: -> [B,128,4,4] ----
  {
    const int N = batch * 4 * 4;
    dim3 grid((N + 63) / 64, 4);
    conv_wmma_kernel<64, 7, 7, 128, 4, 4, 5, 2, 2, 2, 4, _Float16>
        <<<grid, 256, 0, stream>>>(act3, w4h, scale4, shift4, act4, batch);
  }
  // ---- 5) maxpool 2x2 -> seq [B,32,16] f32 ----
  {
    const int total = batch * 512;
    pool_kernel<<<(total + 255) / 256, 256, 0, stream>>>(act4, seqp, batch);
  }
  // ---- 6) LTC cell, one wave per sample ----
  {
    dim3 grid((batch + 7) / 8);
    ltc_kernel<<<grid, 256, 0, stream>>>(seqp, sensory_mu, sensory_sig,
                                         sens_we, sens_weff, mu, sigma, w_we, w_eff,
                                         glp, vleak, cmtp, input_w, input_b,
                                         output_w, output_b, out, batch);
  }
}
